// Model_67388036874385
// MI455X (gfx1250) — matrix operand order repaired and checked by reading, not compiled, not run on hardware
//
#include <hip/hip_runtime.h>
#include <hip/hip_bf16.h>

typedef __attribute__((ext_vector_type(16))) _Float16 v16h;
typedef __attribute__((ext_vector_type(8)))  _Float16 v8h;
typedef __attribute__((ext_vector_type(4)))  _Float16 h4;
typedef __attribute__((ext_vector_type(8)))  float    v8f;
typedef __attribute__((ext_vector_type(4)))  float    v4f;

#define INV_SQRT_D 0.044194173824159216f
#define LDS_STRIDE 40   // halves per LDS tile row (80B: 16B-aligned, pad vs bank conflicts)

// ---------------------------------------------------------------------------
// Tiled WMMA GEMM:  C[M,N] = A[M,K] @ B + bias   (NO ragged edges:
//   requires M % 64 == 0, N % 128 == 0, K % 32 == 0 — guaranteed by caller)
//   bTransposed == 0 : Bm is [K,N] row-major (ldb = row stride)
//   bTransposed == 1 : Bm is [N,K] row-major (weights), element B[n*ldb + k]
// Block: 256 threads (8 waves). Block tile 64(M) x 128(N). Wave tile 32x32
// (2x2 WMMA f32_16x16x32_f16 tiles). K stepped by 32 through LDS with a
// register-staged prefetch pipeline. LDS addressed by integer offsets so the
// compiler keeps ds_store_* (generic pointers would force flat_store_*).
// ---------------------------------------------------------------------------
__global__ __launch_bounds__(256)
void gemm_wmma(const float* __restrict__ A, int lda,
               const float* __restrict__ Bm, int ldb, int bTransposed,
               const float* __restrict__ bias,
               float* __restrict__ C, int ldc,
               int M, int N, int K)
{
    __shared__ __attribute__((aligned(16))) _Float16 As[64 * LDS_STRIDE];
    __shared__ __attribute__((aligned(16))) _Float16 Bs[128 * LDS_STRIDE];

    const int tid   = threadIdx.x;
    const int lane  = tid & 31;
    const int wid   = tid >> 5;
    const int mBase = (wid & 1) * 32;   // wave M offset in block tile
    const int nBase = (wid >> 1) * 32;  // wave N offset in block tile
    const int blockM = blockIdx.y * 64;
    const int blockN = blockIdx.x * 128;
    const int hi16   = (lane >> 4) & 1;

    // ---- precompute per-thread staging addresses (LDS as int offsets) ----
    const float* aSrc[2];
    int          aOff[2];
#pragma unroll
    for (int it = 0; it < 2; ++it) {
        int g = it * 256 + tid;             // 512 float4 groups, 64x32 tile
        int r = g >> 3, kg = (g & 7) * 4;
        aSrc[it] = A + (long)(blockM + r) * lda + kg;
        aOff[it] = r * LDS_STRIDE + kg;
    }
    const float* bSrc[4];
    int          bOff[4];
    long bStep;
    if (bTransposed) {
        // weights [N,K]: contiguous in k
#pragma unroll
        for (int it = 0; it < 4; ++it) {
            int g = it * 256 + tid;         // 1024 groups, 128x32 tile
            int n = g >> 3, kg = (g & 7) * 4;
            bSrc[it] = Bm + (long)(blockN + n) * ldb + kg;
            bOff[it] = n * LDS_STRIDE + kg;
        }
        bStep = 32;
    } else {
        // [K,N]: contiguous in n
#pragma unroll
        for (int it = 0; it < 4; ++it) {
            int g = it * 256 + tid;
            int kk = g >> 5, ng = (g & 31) * 4;
            bSrc[it] = Bm + (long)kk * ldb + blockN + ng;
            bOff[it] = ng * LDS_STRIDE + kk;    // rows ng..ng+3
        }
        bStep = (long)32 * ldb;
    }

    // ---- prologue: load first K-tile into registers ----------------------
    v4f aR[2], bR[4];
#pragma unroll
    for (int it = 0; it < 2; ++it) aR[it] = *(const v4f*)aSrc[it];
#pragma unroll
    for (int it = 0; it < 4; ++it) bR[it] = *(const v4f*)bSrc[it];

    v8f acc[2][2] = {};

    for (int k0 = 0; k0 < K; k0 += 32) {
        // ---- convert staged registers, store to LDS (DS pipe) ------------
#pragma unroll
        for (int it = 0; it < 2; ++it) {
            h4 ph;
#pragma unroll
            for (int i = 0; i < 4; ++i) ph[i] = (_Float16)aR[it][i];
            *(h4*)&As[aOff[it]] = ph;
        }
        if (bTransposed) {
#pragma unroll
            for (int it = 0; it < 4; ++it) {
                h4 ph;
#pragma unroll
                for (int i = 0; i < 4; ++i) ph[i] = (_Float16)bR[it][i];
                *(h4*)&Bs[bOff[it]] = ph;
            }
        } else {
#pragma unroll
            for (int it = 0; it < 4; ++it)
#pragma unroll
                for (int i = 0; i < 4; ++i)
                    Bs[bOff[it] + i * LDS_STRIDE] = (_Float16)bR[it][i];
        }
        __syncthreads();

        // ---- prefetch next K-tile (overlaps fragment loads + WMMA) -------
        if (k0 + 32 < K) {
#pragma unroll
            for (int it = 0; it < 2; ++it) { aSrc[it] += 32;    aR[it] = *(const v4f*)aSrc[it]; }
#pragma unroll
            for (int it = 0; it < 4; ++it) { bSrc[it] += bStep; bR[it] = *(const v4f*)bSrc[it]; }
        }

        // ---- fragments per ISA VGPR layouts ------------------------------
        // A 16-bit 16x32: lanes 0-15 K={0..7,16..23}, lanes 16-31 K={8..15,24..31}
        v16h afrag[2];
#pragma unroll
        for (int mi = 0; mi < 2; ++mi) {
            int rb = (mBase + mi * 16 + (lane & 15)) * LDS_STRIDE;
            v8h lo = *(const v8h*)&As[rb + hi16 * 8];
            v8h hi = *(const v8h*)&As[rb + 16 + hi16 * 8];
            v16h a;
#pragma unroll
            for (int i = 0; i < 8; ++i) { a[i] = lo[i]; a[i + 8] = hi[i]; }
            afrag[mi] = a;
        }
        // B 16-bit 32x16: lanes 0-15 K={0..7,16..23}, lanes 16-31 K={8..15,24..31}
        v16h bfrag[2];
#pragma unroll
        for (int ni = 0; ni < 2; ++ni) {
            int nb = (nBase + ni * 16 + (lane & 15)) * LDS_STRIDE;
            v8h lo = *(const v8h*)&Bs[nb + hi16 * 8];
            v8h hi = *(const v8h*)&Bs[nb + 16 + hi16 * 8];
            v16h bb;
#pragma unroll
            for (int i = 0; i < 8; ++i) { bb[i] = lo[i]; bb[i + 8] = hi[i]; }
            bfrag[ni] = bb;
        }
#pragma unroll
        for (int mi = 0; mi < 2; ++mi)
#pragma unroll
            for (int ni = 0; ni < 2; ++ni)
                acc[mi][ni] = __builtin_amdgcn_wmma_f32_16x16x32_f16(
                    false, afrag[mi], false, bfrag[ni],
                    (short)0, acc[mi][ni], false, false);
        __syncthreads();
    }

    // ---- epilogue: C/D layout lanes0-15 VGPRr<->M=r, lanes16-31 VGPRr<->M=8+r
#pragma unroll
    for (int mi = 0; mi < 2; ++mi)
#pragma unroll
        for (int ni = 0; ni < 2; ++ni) {
            int gcol = blockN + nBase + ni * 16 + (lane & 15);
            float bv = bias ? bias[gcol] : 0.f;
#pragma unroll
            for (int r = 0; r < 8; ++r) {
                int grow = blockM + mBase + mi * 16 + hi16 * 8 + r;
                C[(long)grow * ldc + gcol] = acc[mi][ni][r] + bv;
            }
        }
}

// ---------------------------------------------------------------------------
// Setup A: positional features, x_trunk = relu(b_first), c_kvx, q_c
// ---------------------------------------------------------------------------
__global__ __launch_bounds__(256)
void setup_a(const float* __restrict__ b_first,
             const float* __restrict__ W_kvx, const float* __restrict__ b_kvx,
             const float* __restrict__ W_q,   const float* __restrict__ b_q,
             float* __restrict__ posf, float* __restrict__ xtr,
             float* __restrict__ ckvx, float* __restrict__ qc)
{
    const int t = threadIdx.x;
    for (int p = t; p < 121; p += 256) {
        int r = p / 11, c = p % 11;
        float lr = -1.f + 2.f * r / 10.f;
        float lc = -1.f + 2.f * c / 10.f;
        float xs[3] = { lr, lc, 0.5f * (lr + lc) };
        float per[3] = { 1.f, 2.f, 4.f };
        const float tw = 6.2831853071795864f;
        for (int pi = 0; pi < 3; ++pi)
            for (int a = 0; a < 3; ++a) {
                float ang = tw * xs[a] / per[pi];
                posf[p * 18 + pi * 3 + a]     = cosf(ang);
                posf[p * 18 + 9 + pi * 3 + a] = sinf(ang);
            }
    }
    for (int i = t; i < 512; i += 256) xtr[i] = fmaxf(b_first[i], 0.f);
    __syncthreads();
    for (int i = t; i < 2048; i += 256) {
        float s = b_kvx[i];
        const float* w = W_kvx + (long)i * 512;
        for (int d = 0; d < 512; ++d) s += w[d] * xtr[d];
        ckvx[i] = s;
    }
    for (int i = t; i < 1024; i += 256) {
        float s = b_q[i];
        const float* w = W_q + (long)i * 512;
        for (int d = 0; d < 512; ++d) s += w[d] * xtr[d];
        qc[i] = s;
    }
}

// ---------------------------------------------------------------------------
// Setup B: dot_base[2][128] (k_base[p,h] . q_c[h]) and wdot[2][8]
// ---------------------------------------------------------------------------
__global__ __launch_bounds__(256)
void setup_b(const float* __restrict__ W_kvb, const float* __restrict__ b_kvb,
             const float* __restrict__ posf, const float* __restrict__ ckvx,
             const float* __restrict__ qc,
             float* __restrict__ dotb, float* __restrict__ wdot)
{
    const int t = threadIdx.x;
    {
        int h = t >> 7, p = t & 127;
        float s = 0.f;
        if (p < 121) {
            for (int d = 0; d < 512; ++d) {
                int idx = h * 512 + d;
                float kb = ckvx[idx] + b_kvb[idx];
                const float* wr = W_kvb + (long)idx * 25 + 7;
                for (int tt = 0; tt < 18; ++tt) kb += posf[p * 18 + tt] * wr[tt];
                s += kb * qc[idx];
            }
        }
        dotb[h * 128 + p] = s;
    }
    if (t < 16) {
        int h = t >> 3, j = t & 7;
        float s = 0.f;
        if (j < 7)
            for (int d = 0; d < 512; ++d) {
                int idx = h * 512 + d;
                s += W_kvb[(long)idx * 25 + j] * qc[idx];
            }
        wdot[t] = s;
    }
}

// ---------------------------------------------------------------------------
// Vext[h][pp][d]: rows 0..120 = v_base (const over batch), rows 121..127 =
// obs-columns of W_kvb (attn-weighted-neighborhood correction)
// ---------------------------------------------------------------------------
__global__ __launch_bounds__(256)
void setup_vext(const float* __restrict__ W_kvb, const float* __restrict__ b_kvb,
                const float* __restrict__ posf, const float* __restrict__ ckvx,
                float* __restrict__ Vext)
{
    int idx = blockIdx.x * 256 + threadIdx.x;
    const int total = 2 * 128 * 512;
    for (; idx < total; idx += gridDim.x * 256) {
        int h = idx >> 16;
        int rem = idx & 65535;
        int pp = rem >> 9;
        int d  = rem & 511;
        int row = 1024 + h * 512 + d;
        const float* wr = W_kvb + (long)row * 25;
        float v;
        if (pp < 121) {
            v = ckvx[row] + b_kvb[row];
            for (int tt = 0; tt < 18; ++tt) v += posf[pp * 18 + tt] * wr[7 + tt];
        } else {
            v = wr[pp - 121];
        }
        Vext[idx] = v;
    }
}

// kp[p][d] = b_kp[d] + pos[p] . W_kp[d,:]  ; rows 121..127 zero-padded so the
// policy GEMM can run with N = 128 (no ragged tiles)
__global__ __launch_bounds__(256)
void setup_kp(const float* __restrict__ W_kp, const float* __restrict__ b_kp,
              const float* __restrict__ posf, float* __restrict__ kp)
{
    int idx = blockIdx.x * 256 + threadIdx.x;
    const int total = 128 * 512;
    for (; idx < total; idx += gridDim.x * 256) {
        int p = idx >> 9, d = idx & 511;
        float s = 0.f;
        if (p < 121) {
            s = b_kp[d];
            const float* w = W_kp + d * 18;
            for (int tt = 0; tt < 18; ++tt) s += posf[p * 18 + tt] * w[tt];
        }
        kp[idx] = s;
    }
}

// ---------------------------------------------------------------------------
// Per-batch attention: neighborhoods, logits, softmax, moments -> A1 matrix
// A1[h][b][0..120] = attn, A1[h][b][121..127] = sum_p attn*nb[:,j]
// ---------------------------------------------------------------------------
__global__ __launch_bounds__(128)
void attn_kernel(const float* __restrict__ obs,
                 const float* __restrict__ dotb, const float* __restrict__ wdot,
                 float* __restrict__ A1)
{
    __shared__ float sp[169];     // 13x13 padded board
    __shared__ float red[128];
    __shared__ float msh[16];
    const int t = threadIdx.x;
    const int b = blockIdx.x;

    for (int i = t; i < 169; i += 128) sp[i] = 0.f;
    __syncthreads();
    if (t < 121) {
        int r = t / 11, c = t % 11;
        float o0 = obs[((long)b * 121 + t) * 2 + 0];
        float o1 = obs[((long)b * 121 + t) * 2 + 1];
        sp[(r + 1) * 13 + (c + 1)] = o0 - o1;
    }
    __syncthreads();

    const int drs[7] = { -1, -1, 0, 0, 0, -1, -1 };
    const int dcs[7] = {  0,  1,-1, 0, 1, -1,  0 };
    float nb[7];
    float dts[2];
    if (t < 121) {
        int r = t / 11, c = t % 11;
#pragma unroll
        for (int j = 0; j < 7; ++j)
            nb[j] = sp[(r + 1 + drs[j]) * 13 + (c + 1 + dcs[j])];
#pragma unroll
        for (int h = 0; h < 2; ++h) {
            float s = dotb[h * 128 + t];
#pragma unroll
            for (int j = 0; j < 7; ++j) s += nb[j] * wdot[h * 8 + j];
            dts[h] = s * INV_SQRT_D;
        }
    } else {
#pragma unroll
        for (int j = 0; j < 7; ++j) nb[j] = 0.f;
        dts[0] = dts[1] = -1e30f;
    }

    float attn[2];
    for (int h = 0; h < 2; ++h) {
        red[t] = dts[h]; __syncthreads();
        for (int s = 64; s > 0; s >>= 1) { if (t < s) red[t] = fmaxf(red[t], red[t + s]); __syncthreads(); }
        float mx = red[0]; __syncthreads();
        float e = (t < 121) ? expf(dts[h] - mx) : 0.f;
        red[t] = e; __syncthreads();
        for (int s = 64; s > 0; s >>= 1) { if (t < s) red[t] += red[t + s]; __syncthreads(); }
        float sum = red[0]; __syncthreads();
        attn[h] = e / sum;
    }
    for (int h = 0; h < 2; ++h)
        for (int j = 0; j < 7; ++j) {
            red[t] = attn[h] * nb[j]; __syncthreads();
            for (int s = 64; s > 0; s >>= 1) { if (t < s) red[t] += red[t + s]; __syncthreads(); }
            if (t == 0) msh[h * 8 + j] = red[0];
            __syncthreads();
        }
    for (int h = 0; h < 2; ++h) {
        float v = (t < 121) ? attn[h] : msh[h * 8 + (t - 121)];
        A1[((long)h * 512 + b) * 128 + t] = v;
    }
}

// ---------------------------------------------------------------------------
// Policy finalize: logits = (qq.kx + pd)/sqrt(D), log_softmax over 121
// ---------------------------------------------------------------------------
__global__ __launch_bounds__(128)
void policy_kernel(const float* __restrict__ qq, const float* __restrict__ kx,
                   const float* __restrict__ pd, float* __restrict__ out)
{
    __shared__ float red[128];
    const int t = threadIdx.x;
    const int b = blockIdx.x;
    const int pol = blockIdx.y;
    const float* q = qq + ((long)pol * 512 + b) * 512;
    const float* k = kx + ((long)pol * 512 + b) * 512;
    float part = 0.f;
    for (int i = t; i < 512; i += 128) part += q[i] * k[i];
    red[t] = part; __syncthreads();
    for (int s = 64; s > 0; s >>= 1) { if (t < s) red[t] += red[t + s]; __syncthreads(); }
    float sdot = red[0]; __syncthreads();

    float lg = (t < 121) ? (pd[((long)pol * 512 + b) * 128 + t] + sdot) * INV_SQRT_D : -1e30f;
    red[t] = lg; __syncthreads();
    for (int s = 64; s > 0; s >>= 1) { if (t < s) red[t] = fmaxf(red[t], red[t + s]); __syncthreads(); }
    float mx = red[0]; __syncthreads();
    float e = (t < 121) ? expf(lg - mx) : 0.f;
    red[t] = e; __syncthreads();
    for (int s = 64; s > 0; s >>= 1) { if (t < s) red[t] += red[t + s]; __syncthreads(); }
    float lsum = logf(red[0]);
    if (t < 121) out[(long)pol * (512 * 121) + (long)b * 121 + t] = lg - mx - lsum;
}

// ---------------------------------------------------------------------------
extern "C" void kernel_launch(void* const* d_in, const int* in_sizes, int n_in,
                              void* d_out, int out_size, void* d_ws, size_t ws_size,
                              hipStream_t stream)
{
    const float* obs    = (const float*)d_in[0];
    const float* bfirst = (const float*)d_in[2];
    const float* Wkvx   = (const float*)d_in[3];
    const float* bkvx   = (const float*)d_in[4];
    const float* Wkvb   = (const float*)d_in[5];
    const float* bkvb   = (const float*)d_in[6];
    const float* Wq     = (const float*)d_in[7];
    const float* bq     = (const float*)d_in[8];
    const float* Wfin   = (const float*)d_in[9];
    const float* bfin   = (const float*)d_in[10];
    const float* Wsec   = (const float*)d_in[11];
    const float* bsec   = (const float*)d_in[12];
    const float* Wkp[2] = { (const float*)d_in[13], (const float*)d_in[19] };
    const float* bkp[2] = { (const float*)d_in[14], (const float*)d_in[20] };
    const float* Wkx[2] = { (const float*)d_in[15], (const float*)d_in[21] };
    const float* bkx[2] = { (const float*)d_in[16], (const float*)d_in[22] };
    const float* Wpq[2] = { (const float*)d_in[17], (const float*)d_in[23] };
    const float* bpq[2] = { (const float*)d_in[18], (const float*)d_in[24] };

    float* ws   = (float*)d_ws;
    float* posf = ws + 0;          // 128*18
    float* xtr  = ws + 2304;       // 512
    float* ckvx = ws + 2816;       // 2048
    float* qc   = ws + 4864;       // 1024
    float* dotb = ws + 5888;       // 2*128
    float* wdot = ws + 6144;       // 16
    float* Vext = ws + 6400;       // 2*128*512
    float* kp   = ws + 137472;     // 2*128*512
    float* A1   = ws + 268544;     // 2*512*128
    float* vals = ws + 399616;     // 512*1024
    float* x1   = ws + 923904;     // 512*512
    float* x2   = ws + 1186048;    // 512*512
    float* qqb  = ws + 1448192;    // 2*512*512
    float* kxb  = ws + 1972480;    // 2*512*512
    float* pd   = ws + 2496768;    // 2*512*128   (end: 2627840 floats ~10.5MB)

    setup_a<<<1, 256, 0, stream>>>(bfirst, Wkvx, bkvx, Wq, bq, posf, xtr, ckvx, qc);
    setup_b<<<1, 256, 0, stream>>>(Wkvb, bkvb, posf, ckvx, qc, dotb, wdot);
    setup_vext<<<256, 256, 0, stream>>>(Wkvb, bkvb, posf, ckvx, Vext);
    setup_kp<<<128, 256, 0, stream>>>(Wkp[0], bkp[0], posf, kp);
    setup_kp<<<128, 256, 0, stream>>>(Wkp[1], bkp[1], posf, kp + 128 * 512);

    attn_kernel<<<512, 128, 0, stream>>>(obs, dotb, wdot, A1);

    // vals[:, h*512:(h+1)*512] = A1_h[512,128] @ Vext_h[128,512]
    for (int h = 0; h < 2; ++h)
        gemm_wmma<<<dim3(4, 8), 256, 0, stream>>>(A1 + (long)h * 512 * 128, 128,
                                                  Vext + (long)h * 128 * 512, 512, 0,
                                                  nullptr, vals + h * 512, 1024,
                                                  512, 512, 128);
    // x1 = vals @ W_fin^T + b_fin
    gemm_wmma<<<dim3(4, 8), 256, 0, stream>>>(vals, 1024, Wfin, 1024, 1, bfin,
                                              x1, 512, 512, 512, 1024);
    // x2 = x1 @ W_sec^T + b_sec
    gemm_wmma<<<dim3(4, 8), 256, 0, stream>>>(x1, 512, Wsec, 512, 1, bsec,
                                              x2, 512, 512, 512, 512);
    for (int pol = 0; pol < 2; ++pol) {
        gemm_wmma<<<dim3(4, 8), 256, 0, stream>>>(x2, 512, Wpq[pol], 512, 1, bpq[pol],
                                                  qqb + (long)pol * 512 * 512, 512, 512, 512, 512);
        gemm_wmma<<<dim3(4, 8), 256, 0, stream>>>(x2, 512, Wkx[pol], 512, 1, bkx[pol],
                                                  kxb + (long)pol * 512 * 512, 512, 512, 512, 512);
        // pd = qq @ kp^T   (kp zero-padded to 128 rows -> N = 128, exact tiles)
        gemm_wmma<<<dim3(1, 8), 256, 0, stream>>>(qqb + (long)pol * 512 * 512, 512,
                                                  kp + (long)pol * 128 * 512, 512, 1,
                                                  nullptr, pd + (long)pol * 512 * 128, 128,
                                                  512, 128, 512);
    }
    policy_kernel<<<dim3(512, 2), 128, 0, stream>>>(qqb, kxb, pd, (float*)d_out);
}
